// HLH_block_78151224918229
// MI455X (gfx1250) — matrix operand order repaired and checked by reading, not compiled, not run on hardware
//
#include <hip/hip_runtime.h>
#include <hip/hip_bf16.h>
#include <stdint.h>

// ---------------------------------------------------------------------------
// Types for CDNA5 WMMA (wave32): A/B = 16 bf16 per lane, C/D = 8 f32 per lane.
// ---------------------------------------------------------------------------
typedef __attribute__((ext_vector_type(16))) __bf16 v16bf;
typedef __attribute__((ext_vector_type(8)))  float  v8f;

#define LRELU_SLOPE 0.01f

// Order-preserving float <-> uint encoding so segment-max can use atomicMax(u32).
__device__ __forceinline__ unsigned f2ord(float f) {
  unsigned u = __float_as_uint(f);
  return (u & 0x80000000u) ? ~u : (u | 0x80000000u);
}
__device__ __forceinline__ float ord2f(unsigned e) {
  return (e & 0x80000000u) ? __uint_as_float(e & 0x7fffffffu)
                           : __uint_as_float(~e);
}

// ---------------------------------------------------------------------------
// Weight packing: PB[((n * (K/32) + kb)*32 + kk] = bf16(B[(kb*32+kk)*N + n]).
// A lane needing column n, k = kb*32 + 16*half + i (i=0..15) then reads one
// contiguous, 32B-aligned v16bf -> two global_load_b128, no converts in GEMM.
// ---------------------------------------------------------------------------
__global__ void pack_b_kernel(const float* __restrict__ B, __bf16* __restrict__ PB,
                              int K, int N)
{
  const long long i = (long long)blockIdx.x * blockDim.x + threadIdx.x;
  if (i >= (long long)K * N) return;
  const int n = (int)(i % N), k = (int)(i / N);
  PB[((size_t)n * (K >> 5) + (k >> 5)) * 32 + (k & 31)] = (__bf16)B[i];
}

// Column-concatenated pack of [W0 | W1 | W2] (all K rows) into the same layout.
__global__ void pack_b3_kernel(const float* __restrict__ W0, int N0,
                               const float* __restrict__ W1, int N1,
                               const float* __restrict__ W2, int N2,
                               __bf16* __restrict__ PB, int K)
{
  const int Ntot = N0 + N1 + N2;
  const long long i = (long long)blockIdx.x * blockDim.x + threadIdx.x;
  if (i >= (long long)K * Ntot) return;
  const int n = (int)(i % Ntot), k = (int)(i / Ntot);
  float v;
  if (n < N0)            v = W0[(size_t)k * N0 + n];
  else if (n < N0 + N1)  v = W1[(size_t)k * N1 + (n - N0)];
  else                   v = W2[(size_t)k * N2 + (n - N0 - N1)];
  PB[((size_t)n * (K >> 5) + (k >> 5)) * 32 + (k & 31)] = (__bf16)v;
}

// Fused bias: zeros for the ni/nj column ranges, bnode for the hfeat range.
__global__ void make_bias_kernel(const float* __restrict__ bnode,
                                 float* __restrict__ biasf, int off, int Ntot, int len)
{
  const int i = blockIdx.x * blockDim.x + threadIdx.x;
  if (i >= Ntot) return;
  biasf[i] = (i >= off && i < off + len) ? bnode[i - off] : 0.0f;
}

// Load one 16x32 A fragment (f32 -> bf16 in-register, 4x b128 loads).
__device__ __forceinline__ v16bf load_a_frag(const float* __restrict__ Ar)
{
  const float4 x0 = *(const float4*)(Ar);
  const float4 x1 = *(const float4*)(Ar + 4);
  const float4 x2 = *(const float4*)(Ar + 8);
  const float4 x3 = *(const float4*)(Ar + 12);
  v16bf a;
  a[0]  = (__bf16)x0.x; a[1]  = (__bf16)x0.y; a[2]  = (__bf16)x0.z; a[3]  = (__bf16)x0.w;
  a[4]  = (__bf16)x1.x; a[5]  = (__bf16)x1.y; a[6]  = (__bf16)x1.z; a[7]  = (__bf16)x1.w;
  a[8]  = (__bf16)x2.x; a[9]  = (__bf16)x2.y; a[10] = (__bf16)x2.z; a[11] = (__bf16)x2.w;
  a[12] = (__bf16)x3.x; a[13] = (__bf16)x3.y; a[14] = (__bf16)x3.z; a[15] = (__bf16)x3.w;
  return a;
}

// ---------------------------------------------------------------------------
// GEMM: C[M,N] = A[M,K](f32) @ PB(packed bf16) (+ bias[N]); bf16 WMMA inside.
// One wave owns a 32x64 tile: 2 A fragments share 4 B fragments -> 8 WMMAs
// per K-step at 2 b128 loads per WMMA. M%16==0, K%32==0, N%64==0 always hold;
// an M%32==16 tail tile runs with the second row-fragment disabled
// (wave-uniform flag, EXEC stays all-1s for every WMMA).
// ---------------------------------------------------------------------------
__global__ __launch_bounds__(256)
void gemm_bf16_wmma(const float* __restrict__ A, const __bf16* __restrict__ PB,
                    const float* __restrict__ bias, float* __restrict__ C,
                    int M, int N, int K)
{
  const int lane   = threadIdx.x & 31;
  const int wave   = blockIdx.x * (blockDim.x >> 5) + (threadIdx.x >> 5);
  const int tilesN = N >> 6;                  // strips of 64 columns
  const int tm     = wave / tilesN;           // 32-row tiles
  const int tn     = wave - tm * tilesN;
  const int row0   = tm * 32;
  if (row0 >= M) return;                      // wave-uniform

  const int half  = lane >> 4;                // lane group 0-15 vs 16-31
  const int l15   = lane & 15;
  const int ncol  = (tn << 6) + l15;
  const int kb32s = K >> 5;
  const bool two  = (row0 + 16 < M);          // second 16-row fragment valid?

  const float* Ar0 = A + (size_t)(row0 + l15) * K + 16 * half;
  const float* Ar1 = Ar0 + (size_t)16 * K;

  v8f acc0[4] = {}, acc1[4] = {};
  for (int kb = 0; kb < kb32s; ++kb) {
    const int ko = kb << 5;
    v16bf b[4];
#pragma unroll
    for (int t = 0; t < 4; ++t)
      b[t] = *(const v16bf*)(PB + ((size_t)(ncol + t * 16) * kb32s + kb) * 32
                                 + 16 * half);
    const v16bf a0 = load_a_frag(Ar0 + ko);
#pragma unroll
    for (int t = 0; t < 4; ++t)
      acc0[t] = __builtin_amdgcn_wmma_f32_16x16x32_bf16(
          false, a0, false, b[t], (short)0, acc0[t], false, false);
    if (two) {
      const v16bf a1 = load_a_frag(Ar1 + ko);
#pragma unroll
      for (int t = 0; t < 4; ++t)
        acc1[t] = __builtin_amdgcn_wmma_f32_16x16x32_bf16(
            false, a1, false, b[t], (short)0, acc1[t], false, false);
    }
  }

#pragma unroll
  for (int t = 0; t < 4; ++t) {
    const int n  = ncol + t * 16;
    const float bv = bias ? bias[n] : 0.0f;
#pragma unroll
    for (int r = 0; r < 8; ++r) {             // C/D layout: vgpr r -> row r+8*half
      C[(size_t)(row0 + 8 * half + r) * N + n] = acc0[t][r] + bv;
    }
    if (two) {
#pragma unroll
      for (int r = 0; r < 8; ++r)
        C[(size_t)(row0 + 16 + 8 * half + r) * N + n] = acc1[t][r] + bv;
    }
  }
}

// ---------------------------------------------------------------------------
// Per-edge: f = leaky_relu(f_ni[src] + f_nj[dst] + f_e + bias); writes the
// head-summed edge features and per-head attention logits. One wave per edge.
// fnode is the fused node GEMM output: cols [0,2Fe)=f_ni, [2Fe,4Fe)=f_nj.
// ---------------------------------------------------------------------------
__global__ __launch_bounds__(256)
void egat_edge_kernel(const float* __restrict__ fnode, int Ns,
                      const float* __restrict__ f_e,  const float* __restrict__ bias,
                      const float* __restrict__ attn, const int* __restrict__ src,
                      const int* __restrict__ dst,    float* __restrict__ ef_out,
                      float* __restrict__ logits, int E, int Fe)
{
  const int e    = blockIdx.x * (blockDim.x >> 5) + (threadIdx.x >> 5);
  const int lane = threadIdx.x & 31;
  if (e >= E) return;
  const int s = src[e], d = dst[e];
  const int HFe = Fe << 1;
  const float* ni = fnode + (size_t)s * Ns;
  const float* nj = fnode + (size_t)d * Ns + HFe;
  const float* fe = f_e + (size_t)e * HFe;
  float* eo = ef_out + (size_t)e * Fe;

  for (int h = 0; h < 2; ++h) {
    float lsum = 0.0f;
    for (int f = lane; f < Fe; f += 32) {
      const int idx = h * Fe + f;
      float v = ni[idx] + nj[idx] + fe[idx] + bias[idx];
      v = (v > 0.0f) ? v : LRELU_SLOPE * v;
      lsum += v * attn[idx];
      if (h == 0) eo[f] = v; else eo[f] += v;   // fold heads (reference sums them)
    }
#pragma unroll
    for (int o = 16; o > 0; o >>= 1) lsum += __shfl_xor(lsum, o, 32);
    if (lane == 0) logits[(size_t)e * 2 + h] = lsum;
  }
}

// Segment max over dst via ordered-uint atomicMax. One thread per (edge, head).
__global__ void seg_max_kernel(const float* __restrict__ logits,
                               const int* __restrict__ dst,
                               unsigned* __restrict__ emax, int E)
{
  const long long i = (long long)blockIdx.x * blockDim.x + threadIdx.x;
  if (i >= 2LL * E) return;
  const int e = (int)(i >> 1), h = (int)(i & 1);
  atomicMax(&emax[dst[e] * 2 + h], f2ord(logits[i]));
}

// ex = exp(logit - max[dst]); segment-sum the denominators. In-place on logits.
__global__ void exp_den_kernel(float* __restrict__ logits,
                               const int* __restrict__ dst,
                               const unsigned* __restrict__ emax,
                               float* __restrict__ denom, int E)
{
  const long long i = (long long)blockIdx.x * blockDim.x + threadIdx.x;
  if (i >= 2LL * E) return;
  const int e = (int)(i >> 1), h = (int)(i & 1);
  const int dh = dst[e] * 2 + h;
  const float x = __expf(logits[i] - ord2f(emax[dh]));
  logits[i] = x;
  atomicAdd(&denom[dh], x);
}

// Scatter aggregation: nf_out[dst] += sum_h hfeat[src,h,:] * a[e,h]. Wave/edge.
// hfeat lives in the fused node buffer at column offset 4*Fe, row stride Ns.
__global__ __launch_bounds__(256)
void agg_kernel(const float* __restrict__ fnode, int Ns, int offH,
                const float* __restrict__ ex, const float* __restrict__ denom,
                const int* __restrict__ src, const int* __restrict__ dst,
                float* __restrict__ nf_out, int E, int Fn)
{
  const int e    = blockIdx.x * (blockDim.x >> 5) + (threadIdx.x >> 5);
  const int lane = threadIdx.x & 31;
  if (e >= E) return;
  const int s = src[e], d = dst[e];
  const float a0 = ex[(size_t)e * 2 + 0] / denom[d * 2 + 0];
  const float a1 = ex[(size_t)e * 2 + 1] / denom[d * 2 + 1];
  const float* hf = fnode + (size_t)s * Ns + offH;
  float* out = nf_out + (size_t)d * Fn;
  for (int f = lane; f < Fn; f += 32)
    atomicAdd(&out[f], hf[f] * a0 + hf[Fn + f] * a1);
}

__global__ void fill_f32(float* __restrict__ p, float v, long long n) {
  const long long i = (long long)blockIdx.x * blockDim.x + threadIdx.x;
  if (i < n) p[i] = v;
}
__global__ void fill_u32(unsigned* __restrict__ p, unsigned v, long long n) {
  const long long i = (long long)blockIdx.x * blockDim.x + threadIdx.x;
  if (i < n) p[i] = v;
}
// ef0 = repeat(node_path, 2, axis=0)
__global__ void repeat2_kernel(const float* __restrict__ src, float* __restrict__ dst,
                               long long rows2, int F)
{
  const long long i = (long long)blockIdx.x * blockDim.x + threadIdx.x;
  const long long total = rows2 * F;
  if (i >= total) return;
  const long long r = i / F, f = i - r * F;
  dst[i] = src[(r >> 1) * F + f];
}

// ---------------------------------------------------------------------------
// Host-side orchestration
// ---------------------------------------------------------------------------
static inline int cdiv(long long a, long long b) { return (int)((a + b - 1) / b); }

static void launch_gemm(const float* A, const __bf16* PB, const float* bias,
                        float* C, int M, int N, int K, hipStream_t st)
{
  const int waves  = ((M + 31) / 32) * (N / 64);
  const int blocks = (waves + 7) / 8;                 // 8 waves per 256-thread block
  gemm_bf16_wmma<<<blocks, 256, 0, st>>>(A, PB, bias, C, M, N, K);
}

struct EgatBufs {
  float *node, *fe, *log, *den, *biasf;
  unsigned* emax;
  __bf16 *PBn, *PBe;
};

static void run_egat(const float* nf, const float* ef, const int* src, const int* dst,
                     const float* Wnode, const float* bnode, const float* Wni,
                     const float* Wnj, const float* Wfij, const float* attn,
                     const float* bias, int N, int E, int FinN, int FinE,
                     int Fe, int Fn, float* nf_out, float* ef_out,
                     const EgatBufs& b, hipStream_t st)
{
  const int Ns = 4 * Fe + 2 * Fn;   // fused node-GEMM width (512 or 768)

  // Fused node GEMM: [f_ni | f_nj | hfeat] = nf @ [Wni | Wnj | Wnode] (+bias).
  const long long kn = (long long)FinN * Ns;
  pack_b3_kernel<<<cdiv(kn, 256), 256, 0, st>>>(Wni, 2 * Fe, Wnj, 2 * Fe,
                                                Wnode, 2 * Fn, b.PBn, FinN);
  make_bias_kernel<<<cdiv(Ns, 256), 256, 0, st>>>(bnode, b.biasf, 4 * Fe, Ns, 2 * Fn);
  launch_gemm(nf, b.PBn, b.biasf, b.node, N, Ns, FinN, st);

  // Edge GEMM: f_e = ef @ Wfij.
  const long long ke = (long long)FinE * 2 * Fe;
  pack_b_kernel<<<cdiv(ke, 256), 256, 0, st>>>(Wfij, b.PBe, FinE, 2 * Fe);
  launch_gemm(ef, b.PBe, nullptr, b.fe, E, 2 * Fe, FinE, st);

  const int eb = (E + 7) / 8;
  egat_edge_kernel<<<eb, 256, 0, st>>>(b.node, Ns, b.fe, bias, attn,
                                       src, dst, ef_out, b.log, E, Fe);
  const long long n2 = 2LL * N, e2 = 2LL * E, nfn = (long long)N * Fn;
  fill_u32<<<cdiv(n2, 256), 256, 0, st>>>(b.emax, 0u, n2);
  fill_f32<<<cdiv(n2, 256), 256, 0, st>>>(b.den, 0.0f, n2);
  seg_max_kernel<<<cdiv(e2, 256), 256, 0, st>>>(b.log, dst, b.emax, E);
  exp_den_kernel<<<cdiv(e2, 256), 256, 0, st>>>(b.log, dst, b.emax, b.den, E);
  fill_f32<<<cdiv(nfn, 256), 256, 0, st>>>(nf_out, 0.0f, nfn);
  agg_kernel<<<eb, 256, 0, st>>>(b.node, Ns, 4 * Fe, b.log, b.den,
                                 src, dst, nf_out, E, Fn);
}

extern "C" void kernel_launch(void* const* d_in, const int* in_sizes, int n_in,
                              void* d_out, int out_size, void* d_ws, size_t ws_size,
                              hipStream_t stream)
{
  // Problem sizes (fixed by the reference).
  const int N0 = 50000, E0 = 200000, N1 = 100000, E1 = 400000;
  if (n_in < 38) return;
  if (ws_size < 199666560ull * sizeof(float)) return;

  // Inputs: setup_inputs() insertion order; params flattened in insertion order
  // (l1, l2[0], l2[1], l3, Wlin2, blin2; each egat: Wnode,bnode,Wni,Wnj,Wfij,attn,bias).
  const float* node_feats = (const float*)d_in[0];
  const float* in_np      = (const float*)d_in[2];
  const float* in_ep      = (const float*)d_in[3];
  const float* P[30];
  for (int i = 4; i < 34; ++i) P[i - 4] = (const float*)d_in[i];
  const float* Wlin2 = (const float*)d_in[32];
  const float* blin2 = (const float*)d_in[33];
  const int* src0 = (const int*)d_in[34];
  const int* dst0 = (const int*)d_in[35];
  const int* src1 = (const int*)d_in[36];
  const int* dst1 = (const int*)d_in[37];

  // Workspace carve-up (float elements; all regions >=32B aligned).
  float* ws = (float*)d_ws;
  float* S_node = ws;                        // 51.2M (fused node GEMM out)
  float* S_fe   = S_node + 51200000;         // 51.2M
  float* S_log  = S_fe   + 51200000;         // 0.8M
  unsigned* S_emax = (unsigned*)(S_log + 800000); // 0.2M
  float* S_den  = (float*)(S_emax + 200000); // 0.2M
  float* P_np   = S_den + 200000;            // 12.8M
  float* P_ep   = P_np  + 12800000;          // 25.6M
  float* P_ef0  = P_ep  + 25600000;          // 25.6M
  float* P_ef1  = P_ef0 + 25600000;          // 25.6M
  float* P_nf0  = P_ef1 + 25600000;          // 6.4M
  __bf16* PBn   = (__bf16*)(P_nf0 + 6400000);         // 98304 bf16 (packed fused W)
  __bf16* PBe   = (__bf16*)(P_nf0 + 6400000 + 49152); // 32768 bf16 (packed Wfij/Wlin2)
  float* biasf  = P_nf0 + 6400000 + 49152 + 16384;    // 1024 (fused bias)

  EgatBufs B{S_node, S_fe, S_log, S_den, biasf, S_emax, PBn, PBe};

  // Output regions: [nf | ef | node_path | edge_path]
  float* out    = (float*)d_out;
  float* out_nf = out;                // 50000*128
  float* out_ef = out + 6400000;      // 200000*128
  float* out_np = out + 32000000;     // 100000*128
  float* out_ep = out + 44800000;     // 400000*64

  // Stage 1: line graph, layer l1 (params 0..6).
  run_egat(in_np, in_ep, src1, dst1,
           P[0], P[1], P[2], P[3], P[4], P[5], P[6],
           N1, E1, 128, 64, 64, 128, P_np, P_ep, B, stream);

  // Bond-node features -> directed atom-graph edge features (2 edges per bond).
  repeat2_kernel<<<cdiv((long long)E0 * 128, 256), 256, 0, stream>>>(
      P_np, P_ef0, (long long)E0, 128);

  // Stage 2: atom graph, two l2 layers (params 7..13 and 14..20).
  run_egat(node_feats, P_ef0, src0, dst0,
           P[7], P[8], P[9], P[10], P[11], P[12], P[13],
           N0, E0, 128, 128, 128, 128, P_nf0, P_ef1, B, stream);
  run_egat(P_nf0, P_ef1, src0, dst0,
           P[14], P[15], P[16], P[17], P[18], P[19], P[20],
           N0, E0, 128, 128, 128, 128, out_nf, out_ef, B, stream);

  // x11 = concat(ef[0::2], ef[1::2], axis=1) == ef reinterpreted as [N1, 256].
  pack_b_kernel<<<cdiv(256LL * 128, 256), 256, 0, stream>>>(Wlin2, PBe, 256, 128);
  launch_gemm(out_ef, PBe, blin2, P_np, N1, 128, 256, stream);

  // Stage 3: line graph, layer l3 (params 21..27).
  run_egat(P_np, P_ep, src1, dst1,
           P[21], P[22], P[23], P[24], P[25], P[26], P[27],
           N1, E1, 128, 64, 64, 128, out_np, out_ep, B, stream);

  (void)in_sizes; (void)out_size;
}
